// IDCNN_CRF_50955492000324
// MI455X (gfx1250) — matrix operand order repaired and checked by reading, not compiled, not run on hardware
//
#include <hip/hip_runtime.h>
#include <hip/hip_bf16.h>

typedef __attribute__((ext_vector_type(16))) _Float16 v16h;
typedef __attribute__((ext_vector_type(8)))  _Float16 h8v;
typedef __attribute__((ext_vector_type(8)))  float    v8f;

#define BB   256
#define LL   512
#define FF   128
#define TT   9

// ---------------- embedding gather: f32 emb rows -> f16 activations ----------
__global__ void embed_kernel(const int* __restrict__ inputs,
                             const float* __restrict__ emb,
                             _Float16* __restrict__ x, int total /*B*L*/) {
    int i = blockIdx.x * blockDim.x + threadIdx.x;
    if (i >= total * FF) return;
    int row = i >> 7, c = i & 127;
    int tok = inputs[row];
    x[i] = (_Float16)emb[(size_t)tok * FF + c];
}

// -------- pack a KxN f32 row-major matrix into WMMA B-fragment layout --------
// out layout: [nt][kt][lane(32)][e(16)] halves; lane<16: N=nt*16+lane, K=kt*32+(e<8 ? e : 16+e-8)
//                                          lane>=16: N=nt*16+lane-16, K=kt*32+8+(e<8 ? e : 16+e-8)
__global__ void pack_b_kernel(const float* __restrict__ w, _Float16* __restrict__ out,
                              int K, int N) {
    int idx = blockIdx.x * blockDim.x + threadIdx.x;
    int NT = N >> 4, KT = K >> 5;
    int total = NT * KT * 512;
    if (idx >= total) return;
    int e    = idx & 15;
    int lane = (idx >> 4) & 31;
    int pair = idx >> 9;
    int kt   = pair % KT;
    int nt   = pair / KT;
    int nn   = nt * 16 + (lane & 15);
    int kk   = kt * 32 + ((lane >= 16) ? 8 : 0) + (e & 7) + ((e >= 8) ? 16 : 0);
    out[idx] = (_Float16)w[(size_t)kk * N + nn];
}

// ------- fold linear(F->H) and out(H->T) into W2[128x16], b2[16] (pad T->16) -
__global__ void combine_w_kernel(const float* __restrict__ lin_w,  // 128x256
                                 const float* __restrict__ lin_b,  // 256
                                 const float* __restrict__ out_w,  // 256x9
                                 const float* __restrict__ out_b,  // 9
                                 float* __restrict__ W2,           // 128x16
                                 float* __restrict__ b2) {         // 16
    int n = threadIdx.x;          // 0..15
    int k = blockIdx.x;           // 0..127
    if (n >= 16) return;
    float s = 0.f;
    if (n < TT)
        for (int j = 0; j < 256; ++j) s += lin_w[(size_t)k * 256 + j] * out_w[(size_t)j * TT + n];
    W2[(size_t)k * 16 + n] = s;
    if (k == 0) {
        float bb = 0.f;
        if (n < TT) {
            for (int j = 0; j < 256; ++j) bb += lin_b[j] * out_w[(size_t)j * TT + n];
            bb += out_b[n];
        }
        b2[n] = bb;
    }
}

// ---------------- dilated 3-tap conv as WMMA GEMM, bias + relu, f16 out ------
// x: [B*L,128] f16   wp: packed [s(3)][nt(8)][kt(4)][512] halves   bias: [128] f32
// Input tile (16 rows + dil halo each side) is staged into LDS once per
// workgroup via async global->LDS loads (ASYNCcnt), halo zero-filled for SAME
// padding, so the WMMA loop is branchless and reads A fragments from LDS.
#define LDS_STRIDE 136   // halves per LDS row (128 + 8 pad -> 68 dwords, conflict-free)
__global__ __launch_bounds__(128)
void conv_wmma_kernel(const _Float16* __restrict__ xin,
                      const _Float16* __restrict__ wp,
                      const float* __restrict__ bias,
                      _Float16* __restrict__ xout, int dil) {
    const int NT = 8, KT = 4;
    __shared__ _Float16 sh[20 * LDS_STRIDE];

    int tid  = threadIdx.x;
    int wave = tid >> 5;
    int lane = tid & 31;
    int l0   = blockIdx.x * 16;
    int b    = blockIdx.y;
    int m    = lane & 15;
    bool hi  = lane >= 16;
    int n0   = wave * 32;                   // this wave: n-tiles n0/16 and n0/16+1

    __builtin_prefetch(wp, 0, 0);

    // ---- stage rows [l0-dil, l0+15+dil] into LDS; 16B chunks, zero OOB ----
    int nrows   = 16 + 2 * dil;
    int nchunks = nrows * 16;               // 128 halves/row = 16 x (8-half) chunks
    for (int cid = tid; cid < nchunks; cid += 128) {
        int row = cid >> 4;
        int ch  = cid & 15;
        int l   = l0 - dil + row;
        _Float16* dst = &sh[row * LDS_STRIDE + ch * 8];
        if (l >= 0 && l < LL) {
            const _Float16* src = xin + ((size_t)b * LL + l) * FF + ch * 8;
            unsigned loff = (unsigned)(size_t)dst;          // LDS byte offset
            unsigned long long ga = (unsigned long long)(size_t)src;
            asm volatile("global_load_async_to_lds_b128 %0, %1, off"
                         :: "v"(loff), "v"(ga) : "memory");
        } else {
            h8v z;
            #pragma unroll
            for (int e = 0; e < 8; ++e) z[e] = (_Float16)0;
            *(h8v*)dst = z;
        }
    }
    asm volatile("s_wait_asynccnt 0x0" ::: "memory");
    __syncthreads();

    v8f acc0, acc1;
    #pragma unroll
    for (int r = 0; r < 8; ++r) { acc0[r] = 0.f; acc1[r] = 0.f; }

    int nt0 = n0 >> 4;
    #pragma unroll
    for (int s = 0; s < 3; ++s) {
        int lrow = m + s * dil;             // LDS row for this lane's M, this tap
        #pragma unroll
        for (int kt = 0; kt < KT; ++kt) {
            int kb = kt * 32 + (hi ? 8 : 0);
            const _Float16* ar = &sh[lrow * LDS_STRIDE + kb];
            h8v alo = *(const h8v*)ar;
            h8v ahi = *(const h8v*)(ar + 16);
            v16h a;
            #pragma unroll
            for (int e = 0; e < 8; ++e) { a[e] = alo[e]; a[e + 8] = ahi[e]; }
            v16h b0 = *(const v16h*)(wp + ((((size_t)s * NT + nt0)     * KT + kt) * 512) + lane * 16);
            v16h b1 = *(const v16h*)(wp + ((((size_t)s * NT + nt0 + 1) * KT + kt) * 512) + lane * 16);
            acc0 = __builtin_amdgcn_wmma_f32_16x16x32_f16(false, a, false, b0, (short)0, acc0, false, false);
            acc1 = __builtin_amdgcn_wmma_f32_16x16x32_f16(false, a, false, b1, (short)0, acc1, false, false);
        }
    }

    int col0 = n0 + (lane & 15);
    int col1 = col0 + 16;
    float bi0 = bias[col0], bi1 = bias[col1];
    size_t rbase = (size_t)b * LL + l0;
    #pragma unroll
    for (int r = 0; r < 8; ++r) {
        int M = r + (hi ? 8 : 0);
        float v0 = acc0[r] + bi0; v0 = v0 > 0.f ? v0 : 0.f;
        float v1 = acc1[r] + bi1; v1 = v1 > 0.f ? v1 : 0.f;
        _Float16* orow = xout + (rbase + M) * FF;
        orow[col0] = (_Float16)v0;
        orow[col1] = (_Float16)v1;
    }
}

// ------------- logits = mask(h) @ W2 + b2, WMMA, f32 out [B*L,16] ------------
__global__ __launch_bounds__(128)
void logits_wmma_kernel(const _Float16* __restrict__ h,
                        const _Float16* __restrict__ w2p,  // [kt(4)][512]
                        const float* __restrict__ b2,      // [16]
                        const int* __restrict__ length,
                        float* __restrict__ logits) {
    int wave = threadIdx.x >> 5;
    int lane = threadIdx.x & 31;
    int ltile = blockIdx.x * 4 + wave;
    int b = blockIdx.y;
    int len = length[b]; if (len < 1) len = 1;
    int l0 = ltile * 16;
    int m  = lane & 15;
    bool hi = lane >= 16;
    int l = l0 + m;
    bool ok = l < len;                       // h = where(mask, h, 0)

    v8f acc;
    #pragma unroll
    for (int r = 0; r < 8; ++r) acc[r] = 0.f;

    #pragma unroll
    for (int kt = 0; kt < 4; ++kt) {
        int kb = kt * 32 + (hi ? 8 : 0);
        v16h a;
        #pragma unroll
        for (int e = 0; e < 16; ++e) a[e] = (_Float16)0;
        if (ok) {
            const _Float16* arow = h + ((size_t)b * LL + l) * FF;
            h8v alo = *(const h8v*)(arow + kb);
            h8v ahi = *(const h8v*)(arow + kb + 16);
            #pragma unroll
            for (int e = 0; e < 8; ++e) { a[e] = alo[e]; a[e + 8] = ahi[e]; }
        }
        v16h bf = *(const v16h*)(w2p + (size_t)kt * 512 + lane * 16);
        acc = __builtin_amdgcn_wmma_f32_16x16x32_f16(false, a, false, bf, (short)0, acc, false, false);
    }

    int col = lane & 15;
    float bc = b2[col];
    size_t rbase = (size_t)b * LL + l0;
    #pragma unroll
    for (int r = 0; r < 8; ++r) {
        int M = r + (hi ? 8 : 0);
        logits[(rbase + M) * 16 + col] = acc[r] + bc;
    }
}

// --------- CRF: Viterbi + backtrack + forward logZ + gold, per batch ---------
__global__ void crf_kernel(const float* __restrict__ logits,  // [B*L,16]
                           const int* __restrict__ length,
                           const int* __restrict__ labels,    // [B,L]
                           const float* __restrict__ trans,   // [9,9]
                           const float* __restrict__ startv,
                           const float* __restrict__ endv,
                           float* __restrict__ path_out,      // d_out [B*L]
                           float* __restrict__ per_b,         // ws [B]
                           unsigned char* __restrict__ bp_ws) // [B][L][9]
{
    int b = blockIdx.x * blockDim.x + threadIdx.x;
    if (b >= BB) return;
    float tr[TT * TT];
    for (int i = 0; i < TT * TT; ++i) tr[i] = trans[i];
    int len = length[b]; if (len < 1) len = 1;
    const float* em = logits + (size_t)b * LL * 16;
    float aV[TT], aF[TT];
    for (int j = 0; j < TT; ++j) { float v = startv[j] + em[j]; aV[j] = v; aF[j] = v; }
    unsigned char* bp = bp_ws + (size_t)b * LL * TT;

    for (int t = 1; t < LL; ++t) {
        if (t < len) {
            float nV[TT], nF[TT];
            for (int j = 0; j < TT; ++j) {
                float best = aV[0] + tr[j]; int arg = 0;
                float mx   = aF[0] + tr[j];
                for (int i = 1; i < TT; ++i) {
                    float sv = aV[i] + tr[i * TT + j];
                    if (sv > best) { best = sv; arg = i; }
                    float sf = aF[i] + tr[i * TT + j];
                    if (sf > mx) mx = sf;
                }
                float ssum = 0.f;
                for (int i = 0; i < TT; ++i) ssum += __expf(aF[i] + tr[i * TT + j] - mx);
                float e = em[(size_t)t * 16 + j];
                nV[j] = best + e;
                nF[j] = mx + __logf(ssum) + e;
                bp[(size_t)t * TT + j] = (unsigned char)arg;
            }
            for (int j = 0; j < TT; ++j) { aV[j] = nV[j]; aF[j] = nF[j]; }
        } else {
            for (int j = 0; j < TT; ++j) bp[(size_t)t * TT + j] = (unsigned char)j;
        }
    }
    // last = argmax(alphaV + end)
    float best = aV[0] + endv[0]; int last = 0;
    for (int j = 1; j < TT; ++j) { float v = aV[j] + endv[j]; if (v > best) { best = v; last = j; } }
    // backtrack, masked write
    int cur = last;
    for (int t = LL - 1; t >= 1; --t) {
        path_out[(size_t)b * LL + t] = (float)((t < len) ? cur : 0);
        cur = bp[(size_t)t * TT + cur];
    }
    path_out[(size_t)b * LL] = (float)cur;
    // logZ
    float mx = aF[0] + endv[0];
    for (int j = 1; j < TT; ++j) { float v = aF[j] + endv[j]; if (v > mx) mx = v; }
    float s = 0.f;
    for (int j = 0; j < TT; ++j) s += __expf(aF[j] + endv[j] - mx);
    float logZ = mx + __logf(s);
    // gold score
    const int* lab = labels + (size_t)b * LL;
    float emit_sc = 0.f;
    for (int t = 0; t < len; ++t) emit_sc += em[(size_t)t * 16 + lab[t]];
    float tr_sc = 0.f;
    for (int t = 1; t < len; ++t) tr_sc += tr[lab[t - 1] * TT + lab[t]];
    float gold = emit_sc + tr_sc + startv[lab[0]] + endv[lab[len - 1]];
    per_b[b] = logZ - gold;
}

__global__ void reduce_loss_kernel(const float* __restrict__ per_b, float* __restrict__ out_loss) {
    __shared__ float sh[BB];
    int t = threadIdx.x;
    sh[t] = per_b[t];
    __syncthreads();
    for (int s = BB / 2; s > 0; s >>= 1) {
        if (t < s) sh[t] += sh[t + s];
        __syncthreads();
    }
    if (t == 0) *out_loss = sh[0] / (float)BB;
}

extern "C" void kernel_launch(void* const* d_in, const int* in_sizes, int n_in,
                              void* d_out, int out_size, void* d_ws, size_t ws_size,
                              hipStream_t stream) {
    const int*   inputs  = (const int*)  d_in[0];
    const int*   length  = (const int*)  d_in[1];
    const int*   labels  = (const int*)  d_in[2];
    const float* emb     = (const float*)d_in[3];
    const float* conv0_w = (const float*)d_in[4];
    const float* conv0_b = (const float*)d_in[5];
    const float* convs_w = (const float*)d_in[6];
    const float* convs_b = (const float*)d_in[7];
    const float* lin_w   = (const float*)d_in[8];
    const float* lin_b   = (const float*)d_in[9];
    const float* out_w   = (const float*)d_in[10];
    const float* out_b   = (const float*)d_in[11];
    const float* trans   = (const float*)d_in[12];
    const float* startv  = (const float*)d_in[13];
    const float* endv    = (const float*)d_in[14];
    float* out = (float*)d_out;

    // ---- workspace carve-up (aligned 256B) ----
    char* wsb = (char*)d_ws;
    size_t off = 0;
    auto carve = [&](size_t bytes) -> char* {
        char* p = wsb + off;
        off = (off + bytes + 255) & ~(size_t)255;
        return p;
    };
    _Float16* x0      = (_Float16*)carve((size_t)BB * LL * FF * 2);
    _Float16* x1      = (_Float16*)carve((size_t)BB * LL * FF * 2);
    _Float16* wp0     = (_Float16*)carve((size_t)3 * 16384 * 2);       // conv0 packed
    _Float16* wps     = (_Float16*)carve((size_t)18 * 16384 * 2);      // convs packed
    float*    W2      = (float*)   carve((size_t)FF * 16 * 4);
    float*    b2      = (float*)   carve((size_t)16 * 4);
    _Float16* w2p     = (_Float16*)carve((size_t)4 * 512 * 2);
    float*    logits  = (float*)   carve((size_t)BB * LL * 16 * 4);
    unsigned char* bp = (unsigned char*)carve((size_t)BB * LL * TT);
    float*    per_b   = (float*)   carve((size_t)BB * 4);
    (void)ws_size; (void)in_sizes; (void)n_in; (void)out_size;

    // 1) embedding gather -> f16
    {
        int total = BB * LL * FF;
        embed_kernel<<<(total + 255) / 256, 256, 0, stream>>>(inputs, emb, x0, BB * LL);
    }
    // 2) pack conv weights into WMMA B-fragment layout
    {
        const int elems = 8 * 4 * 512;                     // per 128x128 slice
        for (int k = 0; k < 3; ++k)
            pack_b_kernel<<<(elems + 255) / 256, 256, 0, stream>>>(
                conv0_w + (size_t)k * FF * FF, wp0 + (size_t)k * elems, FF, FF);
        for (int s = 0; s < 18; ++s)
            pack_b_kernel<<<(elems + 255) / 256, 256, 0, stream>>>(
                convs_w + (size_t)s * FF * FF, wps + (size_t)s * elems, FF, FF);
    }
    // 3) fold linear+out: W2 = lin_w@out_w (pad N to 16), b2 = lin_b@out_w+out_b; pack
    combine_w_kernel<<<FF, 16, 0, stream>>>(lin_w, lin_b, out_w, out_b, W2, b2);
    {
        const int elems = 1 * 4 * 512;
        pack_b_kernel<<<(elems + 255) / 256, 256, 0, stream>>>(W2, w2p, FF, 16);
    }
    // 4) conv stack: conv0 (d=1) then 6 layers with dilations (1,1,2,1,1,2)
    {
        dim3 grid(LL / 16, BB);
        conv_wmma_kernel<<<grid, 128, 0, stream>>>(x0, wp0, conv0_b, x1, 1);
        const int dils[6] = {1, 1, 2, 1, 1, 2};
        _Float16* bufs[2] = {x0, x1};
        int cur = 1;
        for (int i = 0; i < 6; ++i) {
            conv_wmma_kernel<<<grid, 128, 0, stream>>>(
                bufs[cur], wps + (size_t)i * 3 * 16384, convs_b + (size_t)i * FF,
                bufs[1 - cur], dils[i]);
            cur = 1 - cur;
        }
        // final h is in bufs[cur] == x1 (6 flips from 1 -> 1)
        // 5) logits = mask(h) @ W2 + b2
        dim3 lgrid(LL / 16 / 4, BB);
        logits_wmma_kernel<<<lgrid, 128, 0, stream>>>(bufs[cur], w2p, b2, length, logits);
    }
    // 6) CRF viterbi + forward + gold per batch
    crf_kernel<<<BB / 64, 64, 0, stream>>>(logits, length, labels, trans, startv, endv,
                                           out, per_b, bp);
    // 7) loss = mean(logZ - gold)
    reduce_loss_kernel<<<1, BB, 0, stream>>>(per_b, out + (size_t)BB * LL);
}
